// Self_Attention_26895085207764
// MI455X (gfx1250) — matrix operand order repaired and checked by reading, not compiled, not run on hardware
//
#include <hip/hip_runtime.h>

// Single-head causal attention, B=2, T=4096, C=1024, fp32 in/out.
// All GEMM stages via v_wmma_f32_16x16x32_bf16 (bf16 operands, fp32 accum).
// Stage 0: one-shot fp32->bf16 conversion of x and weights (native v_cvt).
// Stage 1: QKV projection (2 M-tiles/wave), Q pre-scaled by 1/sqrt(C), V^T stored.
// Stage 2: fused flash attention: 8-wave split-K S via WMMA (hoisted Q frags),
//          LDS reduce + warp-per-row online softmax (m,l in LDS), accumulator
//          rescale, 8 AV WMMAs per 32-key block, final 1/l normalization.
// Stage 3: out = O @ Wo^T + bo (2 M-tiles/wave).
// Workspace: xb 16MB + W*4 bf16 8MB + Q,K,Vt,O 64MB ~= 88 MB.

typedef __attribute__((ext_vector_type(16))) __bf16 v16bf;
typedef __attribute__((ext_vector_type(8)))  float  v8f;

#define C_DIM 1024
#define T_SEQ 4096
#define M_TOT 8192   // B*T

static __device__ __forceinline__ __bf16 f2bf(float f) { return (__bf16)f; }

static __device__ __forceinline__ v8f wmma_bf16(v16bf a, v16bf b, v8f c) {
  // (neg_a, A, neg_b, B, c_mod, C, reuse_a, reuse_b)
  return __builtin_amdgcn_wmma_f32_16x16x32_bf16(false, a, false, b, (short)0, c,
                                                 false, false);
}

// ---- Fragment loaders (wave32 layouts per CDNA5 ISA 7.12.2) -----------------
// A (16x32 MxK): lane L holds row M=L&15; element e -> K = 16*(e>=8) + 8*(L>=16) + (e&7)
static __device__ __forceinline__ v16bf load_a_bf16(const __bf16* base, int ld,
                                                    int row0, int c0, int lane) {
  v16bf a;
  const __bf16* p = base + (long)(row0 + (lane & 15)) * ld + c0;
  int koff = (lane >> 4) << 3;
#pragma unroll
  for (int e = 0; e < 16; ++e)
    a[e] = p[((e & 8) << 1) + koff + (e & 7)];
  return a;
}

// B (32x16 KxN) with B(k,n) = Src[n][k] (row-major NxK source: W, K, or V^T):
// lane L -> col N=L&15; element e -> K = 16*(e>=8) + 8*(L>=16) + (e&7)
static __device__ __forceinline__ v16bf load_bT_bf16(const __bf16* src, long ld,
                                                     int n0, long c0, int lane) {
  v16bf b;
  const __bf16* p = src + (long)(n0 + (lane & 15)) * ld + c0 + ((lane >> 4) << 3);
#pragma unroll
  for (int e = 0; e < 16; ++e) b[e] = p[((e & 8) << 1) + (e & 7)];
  return b;
}

// warp-wide (32 lane) max / sum via xor shuffles
static __device__ __forceinline__ float warp_max(float v) {
#pragma unroll
  for (int mask = 1; mask < 32; mask <<= 1) v = fmaxf(v, __shfl_xor(v, mask, 32));
  return v;
}
static __device__ __forceinline__ float warp_sum(float v) {
#pragma unroll
  for (int mask = 1; mask < 32; mask <<= 1) v += __shfl_xor(v, mask, 32);
  return v;
}

// ---- Kernel 0: one-shot fp32 -> bf16 conversion (native v_cvt path) ----
__global__ void cvt_bf16_kernel(const float* __restrict__ src,
                                __bf16* __restrict__ dst, int n) {
  int i = (blockIdx.x * blockDim.x + threadIdx.x) * 4;
  if (i < n) {
    float4 f = *(const float4*)(src + i);
    dst[i + 0] = f2bf(f.x);
    dst[i + 1] = f2bf(f.y);
    dst[i + 2] = f2bf(f.z);
    dst[i + 3] = f2bf(f.w);
  }
}

// ---- Kernel 1: Q/K/V projections (y = x @ W^T).  Wave: 1 N-tile x 2 M-tiles.
__global__ void qkv_proj_kernel(const __bf16* __restrict__ xb,
                                const __bf16* __restrict__ wq,
                                const __bf16* __restrict__ wk,
                                const __bf16* __restrict__ wv,
                                __bf16* __restrict__ qb,
                                __bf16* __restrict__ kb,
                                __bf16* __restrict__ vt) {
  int lane = threadIdx.x & 31;
  int w    = threadIdx.x >> 5;
  int n0   = (blockIdx.x * 4 + w) * 16;
  int m0   = blockIdx.y * 32;          // two 16-row tiles: m0, m0+16

  v8f cq0 = {}, ck0 = {}, cv0 = {};
  v8f cq1 = {}, ck1 = {}, cv1 = {};
  for (int c0 = 0; c0 < C_DIM; c0 += 32) {
    v16bf a0 = load_a_bf16(xb, C_DIM, m0,      c0, lane);
    v16bf a1 = load_a_bf16(xb, C_DIM, m0 + 16, c0, lane);
    v16bf bq = load_bT_bf16(wq, C_DIM, n0, c0, lane);
    v16bf bk = load_bT_bf16(wk, C_DIM, n0, c0, lane);
    v16bf bv = load_bT_bf16(wv, C_DIM, n0, c0, lane);
    cq0 = wmma_bf16(a0, bq, cq0);  cq1 = wmma_bf16(a1, bq, cq1);
    ck0 = wmma_bf16(a0, bk, ck0);  ck1 = wmma_bf16(a1, bk, ck1);
    cv0 = wmma_bf16(a0, bv, cv0);  cv1 = wmma_bf16(a1, bv, cv1);
  }
  int h = lane >> 4, col = lane & 15;
#pragma unroll
  for (int r = 0; r < 8; ++r) {
    int row0 = m0 + r + 8 * h;
    int row1 = row0 + 16;
    qb[(long)row0 * C_DIM + n0 + col] = f2bf(cq0[r] * 0.03125f); // fold 1/sqrt(C)
    qb[(long)row1 * C_DIM + n0 + col] = f2bf(cq1[r] * 0.03125f);
    kb[(long)row0 * C_DIM + n0 + col] = f2bf(ck0[r]);
    kb[(long)row1 * C_DIM + n0 + col] = f2bf(ck1[r]);
    vt[(long)(n0 + col) * M_TOT + row0] = f2bf(cv0[r]);
    vt[(long)(n0 + col) * M_TOT + row1] = f2bf(cv1[r]);
  }
}

// ---- Kernel 2: fused flash attention: O = softmax(QK^T) @ V ----
// 8 waves/WG per 16-row query tile; wave w owns C/dim slice [128w,128w+128).
// Online softmax: warp w owns rows {w, w+8}; m,l live in LDS.
__global__ void flash_attn_kernel(const __bf16* __restrict__ qb,
                                  const __bf16* __restrict__ kb,
                                  const __bf16* __restrict__ vt,
                                  __bf16* __restrict__ ob) {
  __shared__ float sp[8][16][32];   // per-wave partial S (16 q x 32 k)
  __shared__ float psm[16][32];     // unnormalized P = exp(s - m_new)
  __shared__ float mrow[16], lrow[16], srow[16];

  int tid  = threadIdx.x;
  int lane = tid & 31;
  int w    = tid >> 5;
  int qt   = blockIdx.x;
  int b    = blockIdx.y;
  int q0l  = qt * 16;
  int q0g  = b * T_SEQ + q0l;
  int h = lane >> 4, col = lane & 15;
  int dbase = w * 128;

  if (tid < 16) { mrow[tid] = -1e30f; lrow[tid] = 0.0f; }  // visible after 1st sync

  // loop-invariant Q fragments for this wave's C-slice (4 x 8 VGPRs)
  v16bf aq[4];
#pragma unroll
  for (int cc = 0; cc < 4; ++cc)
    aq[cc] = load_a_bf16(qb, C_DIM, q0g, dbase + cc * 32, lane);

  v8f z = {};
  v8f o[8];
#pragma unroll
  for (int d = 0; d < 8; ++d) o[d] = z;

  int nkb = (qt >> 1) + 1;            // 32-key blocks covering [0, 16*qt+15]
  for (int kblk = 0; kblk < nkb; ++kblk) {
    int k0l = kblk * 32;
    // (1) partial S over this wave's 128-wide C slice
#pragma unroll
    for (int jt = 0; jt < 2; ++jt) {
      v8f acc = {};
      int k0g = b * T_SEQ + k0l + jt * 16;
#pragma unroll
      for (int cc = 0; cc < 4; ++cc) {
        v16bf bk = load_bT_bf16(kb, C_DIM, k0g, dbase + cc * 32, lane);
        acc = wmma_bf16(aq[cc], bk, acc);
      }
#pragma unroll
      for (int r = 0; r < 8; ++r)
        sp[w][r + 8 * h][jt * 16 + col] = acc[r];
    }
    __syncthreads();
    // (2) warp w: reduce + online softmax for rows {w, w+8} (one key/lane)
#pragma unroll
    for (int half = 0; half < 2; ++half) {
      int row = w + 8 * half;
      float s = 0.0f;
#pragma unroll
      for (int ww = 0; ww < 8; ++ww) s += sp[ww][row][lane];
      bool ok = (k0l + lane) <= (q0l + row);          // causal mask
      float mx = warp_max(ok ? s : -1e30f);           // finite: key k0l always valid
      float m_old = mrow[row];
      float mn = fmaxf(m_old, mx);
      float p = ok ? __expf(s - mn) : 0.0f;
      float sum = warp_sum(p);
      psm[row][lane] = p;
      if (lane == 0) {
        float sc = __expf(m_old - mn);                // accumulator rescale factor
        srow[row] = sc;
        lrow[row] = lrow[row] * sc + sum;
        mrow[row] = mn;
      }
    }
    __syncthreads();
    // (3) rescale accumulators, then AV: P(16x32) @ V(32x128-slice), 8 WMMAs
    float sc[8];
#pragma unroll
    for (int r = 0; r < 8; ++r) sc[r] = srow[r + 8 * h];
#pragma unroll
    for (int d = 0; d < 8; ++d)
#pragma unroll
      for (int r = 0; r < 8; ++r) o[d][r] *= sc[r];

    v16bf pa;
    {
      int rowA = lane & 15;
      int koff = (lane >> 4) << 3;
#pragma unroll
      for (int e = 0; e < 16; ++e)
        pa[e] = f2bf(psm[rowA][((e & 8) << 1) + koff + (e & 7)]);
    }
    long ktok = (long)b * T_SEQ + k0l;
#pragma unroll
    for (int d = 0; d < 8; ++d) {
      v16bf bv = load_bT_bf16(vt, M_TOT, dbase + d * 16, ktok, lane);
      o[d] = wmma_bf16(pa, bv, o[d]);
    }
    __syncthreads();
  }
  // final normalization by 1/l and store O tile (bf16)
  if (tid < 16) srow[tid] = 1.0f / lrow[tid];
  __syncthreads();
  float inv[8];
#pragma unroll
  for (int r = 0; r < 8; ++r) inv[r] = srow[r + 8 * h];
#pragma unroll
  for (int d = 0; d < 8; ++d)
#pragma unroll
    for (int r = 0; r < 8; ++r) {
      long row = q0g + r + 8 * h;
      ob[row * C_DIM + dbase + d * 16 + col] = f2bf(o[d][r] * inv[r]);
    }
}

// ---- Kernel 3: out = O @ Wo^T + bo (fp32 out).  Wave: 1 N-tile x 2 M-tiles.
__global__ void out_proj_kernel(const __bf16* __restrict__ ob,
                                const __bf16* __restrict__ wo,
                                const float* __restrict__ bo,
                                float* __restrict__ out) {
  int lane = threadIdx.x & 31;
  int w    = threadIdx.x >> 5;
  int n0   = (blockIdx.x * 4 + w) * 16;
  int m0   = blockIdx.y * 32;

  v8f c0acc = {}, c1acc = {};
  for (int c0 = 0; c0 < C_DIM; c0 += 32) {
    v16bf a0 = load_a_bf16(ob, C_DIM, m0,      c0, lane);
    v16bf a1 = load_a_bf16(ob, C_DIM, m0 + 16, c0, lane);
    v16bf b  = load_bT_bf16(wo, C_DIM, n0, c0, lane);
    c0acc = wmma_bf16(a0, b, c0acc);
    c1acc = wmma_bf16(a1, b, c1acc);
  }
  int h = lane >> 4, col = lane & 15;
  float bias = bo[n0 + col];
#pragma unroll
  for (int r = 0; r < 8; ++r) {
    long row0 = m0 + r + 8 * h;
    out[row0 * C_DIM + n0 + col]        = c0acc[r] + bias;
    out[(row0 + 16) * C_DIM + n0 + col] = c1acc[r] + bias;
  }
}

extern "C" void kernel_launch(void* const* d_in, const int* in_sizes, int n_in,
                              void* d_out, int out_size, void* d_ws, size_t ws_size,
                              hipStream_t stream) {
  (void)in_sizes; (void)n_in; (void)out_size; (void)ws_size;
  const float* x  = (const float*)d_in[0];
  const float* Wq = (const float*)d_in[1];
  const float* Wk = (const float*)d_in[2];
  const float* Wv = (const float*)d_in[3];
  const float* Wo = (const float*)d_in[4];
  const float* bo = (const float*)d_in[5];
  float* out = (float*)d_out;

  const size_t NX = (size_t)M_TOT * C_DIM;   // 8M elements
  const size_t NW = (size_t)C_DIM * C_DIM;   // 1M elements

  // Workspace layout (~88 MB)
  __bf16* xb = (__bf16*)d_ws;
  __bf16* wq = xb + NX;
  __bf16* wk = wq + NW;
  __bf16* wv = wk + NW;
  __bf16* wo = wv + NW;
  __bf16* qb = wo + NW;
  __bf16* kb = qb + NX;
  __bf16* vt = kb + NX;
  __bf16* ob = vt + NX;

  // Stage 0: one-shot bf16 conversion (4 elems/thread)
  cvt_bf16_kernel<<<(int)(NX / 4 / 256), 256, 0, stream>>>(x,  xb, (int)NX);
  cvt_bf16_kernel<<<(int)(NW / 4 / 256), 256, 0, stream>>>(Wq, wq, (int)NW);
  cvt_bf16_kernel<<<(int)(NW / 4 / 256), 256, 0, stream>>>(Wk, wk, (int)NW);
  cvt_bf16_kernel<<<(int)(NW / 4 / 256), 256, 0, stream>>>(Wv, wv, (int)NW);
  cvt_bf16_kernel<<<(int)(NW / 4 / 256), 256, 0, stream>>>(Wo, wo, (int)NW);

  qkv_proj_kernel<<<dim3(16, M_TOT / 32), 128, 0, stream>>>(xb, wq, wk, wv, qb, kb, vt);
  flash_attn_kernel<<<dim3(T_SEQ / 16, 2), 256, 0, stream>>>(qb, kb, vt, ob);
  out_proj_kernel<<<dim3(16, M_TOT / 32), 128, 0, stream>>>(ob, wo, bo, out);
}
